// XEyTransformerLayer_17076789969029
// MI455X (gfx1250) — compile-verified, matrix operand order checked by reading, not run on hardware
//
#include <hip/hip_runtime.h>

// ---------------- constants ----------------
#define C_BS 8
#define C_N  128
#define C_DX 256
#define C_DE 64
#define C_NH 8
#define NE   (C_BS*C_N*C_N)   // 131072 edge rows
#define NN_  (C_BS*C_N)       // 1024 node rows

typedef __attribute__((ext_vector_type(16))) __bf16 v16bf;
typedef __attribute__((ext_vector_type(8)))  float  v8f;

__device__ __forceinline__ __bf16 f2bf(float f) {
  union { float f; unsigned u; } x; x.f = f;
  unsigned r = x.u + 0x7FFFu + ((x.u >> 16) & 1u);
  union { unsigned short s; __bf16 b; } y; y.s = (unsigned short)(r >> 16);
  return y.b;
}

// ---------------- generic batched WMMA GEMM ----------------
// C[bz] = act( A[bz] @ B[bz] + bias ), bf16 staged in LDS in WMMA fragment
// order (k permuted: swap bits 3<->4) so each lane loads its 16 bf16 operand
// values as one contiguous 32B chunk (2x ds_load_b128).
// Fast path: fully-interior tiles staged with unguarded float4 loads
// (global_load_b128) and contiguous 4xbf16 LDS stores.
// aOff = bz*sA ; bOff = (bz/bdiv)*sB ; cOff = (bz/bdiv)*sC + (bz%bdiv)*sC2
#define GBM 64
#define GBN 64
#define GBK 32
__device__ __forceinline__ int kperm(int c) {
  return (c & 7) | ((c & 16) >> 1) | ((c & 8) << 1);
}
__global__ __launch_bounds__(256) void k_gemm(
    const float* __restrict__ A, long long sA, int lda,
    const float* __restrict__ B, long long sB, int ldb, int transB,
    const float* __restrict__ bias,
    float* __restrict__ C, long long sC, long long sC2, int bdiv, int ldc,
    int M, int Nn, int K, int relu)
{
  __shared__ __bf16 As[GBM][GBK];   // [row][perm(k)]
  __shared__ __bf16 Bs[GBN][GBK];   // [col][perm(k)]
  int tid  = threadIdx.x;
  int bz   = blockIdx.z;
  long long aOff = (long long)bz * sA;
  long long bOff = (long long)(bz / bdiv) * sB;
  long long cOff = (long long)(bz / bdiv) * sC + (long long)(bz % bdiv) * sC2;
  int m0 = blockIdx.y * GBM;
  int n0 = blockIdx.x * GBN;
  int wave = tid >> 5, lane = tid & 31;
  int wm = wave >> 1;          // 0..3 : 16-row strip
  int wn = wave &  1;          // 0..1 : 32-col strip
  int half = lane >> 4, l16 = lane & 15;
  int arow  = wm * 16 + l16;
  int bcol0 = wn * 32 + l16;
  int bcol1 = wn * 32 + 16 + l16;
  v8f acc0 = {0.f,0.f,0.f,0.f,0.f,0.f,0.f,0.f};
  v8f acc1 = {0.f,0.f,0.f,0.f,0.f,0.f,0.f,0.f};

  bool fullM = (m0 + GBM) <= M;
  bool fullN = (n0 + GBN) <= Nn;
  bool vecA  = fullM && ((lda & 3) == 0);
  bool vecB  = fullN && ((ldb & 3) == 0);

  for (int k0 = 0; k0 < K; k0 += GBK) {
    bool fullK = (k0 + GBK) <= K;
    // ---- stage A ----
    if (vecA && fullK) {
      for (int x = tid; x < (GBM * GBK) / 4; x += 256) {
        int r  = x >> 3;
        int c4 = (x & 7) * 4;
        float4 v = *(const float4*)(A + aOff + (long long)(m0 + r) * lda + k0 + c4);
        __bf16* dst = &As[r][kperm(c4)];
        dst[0] = f2bf(v.x); dst[1] = f2bf(v.y); dst[2] = f2bf(v.z); dst[3] = f2bf(v.w);
      }
    } else {
      for (int x = tid; x < GBM * GBK; x += 256) {
        int r = x >> 5, c = x & 31;
        int gr = m0 + r, gc = k0 + c;
        float v = (gr < M && gc < K) ? A[aOff + (long long)gr * lda + gc] : 0.f;
        As[r][kperm(c)] = f2bf(v);
      }
    }
    // ---- stage B ----
    if (vecB && fullK) {
      if (transB) {
        for (int x = tid; x < (GBK * GBN) / 4; x += 256) {
          int n  = x >> 3;
          int k4 = (x & 7) * 4;
          float4 v = *(const float4*)(B + bOff + (long long)(n0 + n) * ldb + k0 + k4);
          __bf16* dst = &Bs[n][kperm(k4)];
          dst[0] = f2bf(v.x); dst[1] = f2bf(v.y); dst[2] = f2bf(v.z); dst[3] = f2bf(v.w);
        }
      } else {
        for (int x = tid; x < (GBK * GBN) / 4; x += 256) {
          int k  = x >> 4;
          int n4 = (x & 15) * 4;
          float4 v = *(const float4*)(B + bOff + (long long)(k0 + k) * ldb + n0 + n4);
          int p = kperm(k);
          Bs[n4 + 0][p] = f2bf(v.x);
          Bs[n4 + 1][p] = f2bf(v.y);
          Bs[n4 + 2][p] = f2bf(v.z);
          Bs[n4 + 3][p] = f2bf(v.w);
        }
      }
    } else {
      for (int x = tid; x < GBK * GBN; x += 256) {
        int k = x >> 6, n = x & 63;
        int gk = k0 + k, gn = n0 + n;
        float v;
        if (transB) v = (gk < K && gn < Nn) ? B[bOff + (long long)gn * ldb + gk] : 0.f;
        else        v = (gk < K && gn < Nn) ? B[bOff + (long long)gk * ldb + gn] : 0.f;
        Bs[n][kperm(k)] = f2bf(v);
      }
    }
    __syncthreads();
    v16bf af  = *(const v16bf*)(&As[arow][half * 16]);
    v16bf bf0 = *(const v16bf*)(&Bs[bcol0][half * 16]);
    v16bf bf1 = *(const v16bf*)(&Bs[bcol1][half * 16]);
    acc0 = __builtin_amdgcn_wmma_f32_16x16x32_bf16(false, af, false, bf0,
                                                   (short)0, acc0, false, false);
    acc1 = __builtin_amdgcn_wmma_f32_16x16x32_bf16(false, af, false, bf1,
                                                   (short)0, acc1, false, false);
    __syncthreads();
  }
#pragma unroll
  for (int r = 0; r < 8; r++) {
    int mi = m0 + wm * 16 + r + 8 * half;
    if (mi >= M) continue;
    int ni0 = n0 + wn * 32 + l16;
    int ni1 = ni0 + 16;
    if (ni0 < Nn) {
      float v = acc0[r];
      if (bias) v += bias[ni0];
      if (relu) v = fmaxf(v, 0.f);
      C[cOff + (long long)mi * ldc + ni0] = v;
    }
    if (ni1 < Nn) {
      float v = acc1[r];
      if (bias) v += bias[ni1];
      if (relu) v = fmaxf(v, 0.f);
      C[cOff + (long long)mi * ldc + ni1] = v;
    }
  }
}

// ---------------- small dense linear (tiny M, y-path) ----------------
__global__ void k_small_lin(const float* __restrict__ A, int lda,
                            const float* __restrict__ W,
                            const float* __restrict__ bias,
                            float* __restrict__ C, int ldc,
                            int M, int Nn, int K, int relu, int accum)
{
  int t = blockIdx.x * 256 + threadIdx.x;
  if (t >= M * Nn) return;
  int m = t / Nn, n = t % Nn;
  float acc = bias ? bias[n] : 0.f;
  for (int k = 0; k < K; k++) acc += A[m * lda + k] * W[k * Nn + n];
  if (relu) acc = fmaxf(acc, 0.f);
  if (accum) C[m * ldc + n] += acc; else C[m * ldc + n] = acc;
}

// ---------------- elementwise / masks ----------------
__global__ void k_maskx(float* p, const int* m, int C, int total) {
  int g = blockIdx.x * 256 + threadIdx.x; if (g >= total) return;
  int r = g / C;
  p[g] *= (m[r] ? 1.f : 0.f);
}
__global__ void k_maske(float* p, const int* m, int C) {
  int g = blockIdx.x * 256 + threadIdx.x; if (g >= NE * C) return;
  int t = g / C; int j = t & 127; int i = (t >> 7) & 127; int b = t >> 14;
  p[g] *= ((m[b*128+i] && m[b*128+j]) ? 1.f : 0.f);
}

__global__ void k_prep_pos(const float* pos, const int* nm,
                           float* posm, float* npos, float* normp) {
  int t = blockIdx.x * 256 + threadIdx.x; if (t >= NN_) return;
  float m = nm[t] ? 1.f : 0.f;
  float x = pos[t*3] * m, y = pos[t*3+1] * m, z = pos[t*3+2] * m;
  posm[t*3] = x; posm[t*3+1] = y; posm[t*3+2] = z;
  float nr = sqrtf(x*x + y*y + z*z);
  normp[t] = nr;
  float inv = 1.f / (nr + 1e-7f);
  npos[t*3] = x*inv; npos[t*3+1] = y*inv; npos[t*3+2] = z*inv;
}

__global__ void k_pinfo(const float* posm, const float* npos, float* pinfo) {
  int t = blockIdx.x * 256 + threadIdx.x; if (t >= NE) return;
  int j = t & 127; int i = (t >> 7) & 127; int b = t >> 14;
  int bi = b*128+i, bj = b*128+j;
  float dx = posm[bi*3]-posm[bj*3], dy = posm[bi*3+1]-posm[bj*3+1], dz = posm[bi*3+2]-posm[bj*3+2];
  pinfo[t*2]   = sqrtf(dx*dx + dy*dy + dz*dz + 1e-12f);
  pinfo[t*2+1] = npos[bi*3]*npos[bj*3] + npos[bi*3+1]*npos[bj*3+1] + npos[bi*3+2]*npos[bj*3+2];
}

__global__ void k_dist1(const float* pinfo, const float* normp, const int* nm,
                        const float* w1, const float* b1,
                        const float* wn1, const float* bn1,
                        const float* wn2, const float* bn2, float* dist1) {
  int g = blockIdx.x * 256 + threadIdx.x; if (g >= NE * C_DE) return;
  int c = g & 63; int t = g >> 6;
  int j = t & 127; int i = (t >> 7) & 127; int b = t >> 14;
  float ee = (nm[b*128+i] && nm[b*128+j]) ? 1.f : 0.f;
  float pd = pinfo[t*2], cs = pinfo[t*2+1];
  float ni_ = normp[b*128+i], nj_ = normp[b*128+j];
  float v = pd*w1[c] + cs*w1[64+c] + b1[c] + ni_*wn1[c] + bn1[c] + nj_*wn2[c] + bn2[c];
  dist1[g] = fmaxf(v, 0.f) * ee;
}

__global__ void k_combineE(const float* Y0, const float* xm1, const float* xm2,
                           const float* dA, const float* dM,
                           const float* yadd, const float* ymul,
                           const int* nm, float* Y, float* E2) {
  int g = blockIdx.x * 256 + threadIdx.x; if (g >= NE * C_DE) return;
  int c = g & 63; int t = g >> 6;
  int j = t & 127; int i = (t >> 7) & 127; int b = t >> 14;
  float ee = (nm[b*128+i] && nm[b*128+j]) ? 1.f : 0.f;
  float yv = Y0[g] * xm1[(b*128+j)*64+c] * xm2[(b*128+i)*64+c] * ee;
  yv = (yv + dA[g] + yv * dM[g]) * ee;
  float e2 = (yv + yadd[b*64+c] + yv * ymul[b*64+c]) * ee;
  Y[g] = yv; E2[g] = e2;
}

__global__ void k_make_qh(const float* Q, const float* wa, float* Qh, float scale) {
  int g = blockIdx.x * 256 + threadIdx.x;
  if (g >= C_BS * C_NH * C_N * C_DX) return;
  int d = g & 255; int t = g >> 8;
  int i = t & 127; t >>= 7; int h = t & 7; int b = t >> 3;
  Qh[g] = Q[(long long)(b*128+i)*256 + d] * wa[d*8+h] * scale;
}

__global__ void k_score_self(const float* base, const float* eam, const float* pam,
                             const int* nm, float* s) {
  int g = blockIdx.x * 256 + threadIdx.x; if (g >= 64 * 16384) return;
  int z = g >> 14; int ij = g & 16383;
  int b = z >> 3; int h = z & 7; int i = ij >> 7; int j = ij & 127;
  float ee = (nm[b*128+i] && nm[b*128+j]) ? 1.f : 0.f;
  int t = (b*128+i)*128 + j;
  float a = base[g] * ee;
  a = a * (1.f + eam[t*8+h]);
  a = a * (1.f + pam[t*8+h]) * ee;
  s[g] = a;
}

__global__ void k_wsum(const float* w, float* ws) {
  int h = threadIdx.x; if (h >= 8) return;
  float s = 0.f;
  for (int d = 0; d < 256; d++) s += w[d*8+h];
  ws[h] = s;
}

__global__ void k_cross_max(const float* Qc, const float* Kc, float* mx, float scale) {
  int bi = blockIdx.x; int j = threadIdx.x;   // 1024 blocks, 128 threads
  __shared__ float q[256];
  q[j] = Qc[(long long)bi*256 + j];
  q[j+128] = Qc[(long long)bi*256 + j + 128];
  __syncthreads();
  int b = bi >> 7;
  const float* kr = Kc + (long long)(b*128+j)*256;
  float m = -1e30f;
  for (int d = 0; d < 256; d++) m = fmaxf(m, q[d] * kr[d] * scale);
  mx[(long long)bi*128 + j] = m;
}

__global__ void k_score_cross(const float* basec, const float* mx,
                              const float* wsum, const int* pmm, float* s) {
  int g = blockIdx.x * 256 + threadIdx.x; if (g >= 64 * 16384) return;
  int z = g >> 14; int ij = g & 16383;
  int b = z >> 3; int h = z & 7; int i = ij >> 7; int j = ij & 127;
  int t = (b*128+i)*128 + j;
  float pmi = pmm[b*128+i] ? 1.f : 0.f, pmj = pmm[b*128+j] ? 1.f : 0.f;
  s[g] = (basec[g] - mx[t] * wsum[h]) * pmi * pmj;
}

__global__ void k_softmax(float* s, const int* mask) {
  // rows = 64*128 ; s layout [z][i][j]
  __shared__ float sh[128];
  int row = blockIdx.x; int j = threadIdx.x;
  int z = row >> 7; int b = z >> 3;
  float msk = mask[b*128+j] ? 1.f : 0.f;
  float* p = s + (long long)row * 128;
  float v = (msk > 0.f) ? p[j] : -1e9f;
  sh[j] = v; __syncthreads();
  for (int o = 64; o > 0; o >>= 1) { if (j < o) sh[j] = fmaxf(sh[j], sh[j+o]); __syncthreads(); }
  float mxv = sh[0]; __syncthreads();
  float e = __expf(v - mxv);
  sh[j] = e; __syncthreads();
  for (int o = 64; o > 0; o >>= 1) { if (j < o) sh[j] += sh[j+o]; __syncthreads(); }
  float sum = sh[0];
  p[j] = e / sum;
}

// eto_x : masked moments over j of [b,i,j,64] -> [b,i,256]
__global__ void k_reduce_x(const float* T, const int* nm, float* R) {
  int bi = blockIdx.x; int c = threadIdx.x;  // 1024 blocks, 64 threads
  int b = bi >> 7;
  float nv = 0.f;
  for (int j = 0; j < 128; j++) nv += nm[b*128+j] ? 1.f : 0.f;
  float s1 = 0.f, s2 = 0.f, mn = 1e30f, mx = -1e30f;
  const float* base = T + (long long)bi * 128 * 64;
  for (int j = 0; j < 128; j++) {
    float msk = nm[b*128+j] ? 1.f : 0.f; float inv = 1.f - msk;
    float v = base[j*64 + c];
    s1 += v; s2 += v * v * msk;
    mn = fminf(mn, v + 100000.f * inv);
    mx = fmaxf(mx, v - 100000.f * inv);
  }
  float m = s1 / nv, sd = s2 / nv - m * m;
  float* r = R + (long long)bi * 256;
  r[c] = m; r[64+c] = mn; r[128+c] = mx; r[192+c] = sd;
}

// eto_y : masked moments over (i,j) of [b,i,j,64] -> [b,256]
__global__ void k_reduce_y(const float* T, const int* nm, float* R) {
  int b = blockIdx.x; int c = threadIdx.x;  // 8 blocks, 64 threads
  float nv = 0.f, s1 = 0.f, s2 = 0.f, mn = 1e30f, mx = -1e30f;
  for (int i = 0; i < 128; i++) {
    float mi_ = nm[b*128+i] ? 1.f : 0.f;
    for (int j = 0; j < 128; j++) {
      float ee = mi_ * (nm[b*128+j] ? 1.f : 0.f);
      float inv = 1.f - ee;
      float v = T[(((long long)b*128 + i)*128 + j)*64 + c];
      s1 += v; s2 += v * v * ee; nv += ee;
      mn = fminf(mn, v + 100000.f * inv);
      mx = fmaxf(mx, v - 100000.f * inv);
    }
  }
  float m = s1 / nv, sd = s2 / nv - m * m;
  float* r = R + b * 256;
  r[c] = m; r[64+c] = mn; r[128+c] = mx; r[192+c] = sd;
}

// x_to_y : masked moments over i of newX [b,i,256] -> [b,1024] (newX not pre-masked)
__global__ void k_reduce_x2y(const float* Xn, const int* nm, float* R) {
  int b = blockIdx.x; int c = threadIdx.x;  // 8 blocks, 256 threads
  float nv = 0.f, s1 = 0.f, s1m = 0.f, s2 = 0.f, mn = 1e30f, mx = -1e30f;
  for (int i = 0; i < 128; i++) {
    float m_ = nm[b*128+i] ? 1.f : 0.f; float inv = 1.f - m_;
    float v = Xn[((long long)b*128 + i)*256 + c];
    s1 += v; s1m += v * m_; s2 += v * v * m_; nv += m_;
    mn = fminf(mn, v + 100000.f * inv);
    mx = fmaxf(mx, v - 100000.f * inv);
  }
  float m = s1 / nv;
  float sd = (s2 - 2.f * m * s1m + m * m * nv) / nv;
  float* r = R + b * 1024;
  r[c] = m; r[256+c] = mn; r[512+c] = mx; r[768+c] = sd;
}

__global__ void k_newx(const float* wV, const float* wVc, const int* pmm,
                       const float* exm, const float* pxm,
                       const float* yxm, const float* yxa, float* newX) {
  int g = blockIdx.x * 256 + threadIdx.x; if (g >= NN_ * C_DX) return;
  int r = g >> 8; int c = g & 255; int b = r >> 7;
  float pmv = pmm[r] ? 1.f : 0.f;
  float w = wV[g] * (1.f - pmv) + wVc[g] * pmv;
  w = w * (1.f + exm[g]);
  w = w * (1.f + pxm[g]);
  newX[g] = w * (yxm[b*256+c] + 1.f) + yxa[b*256+c];
}

// residual-fused LayerNorm: out = LN(A + Badd) ; blockDim == C (64 or 256)
__global__ void k_ln(const float* A, const float* Badd, const float* g,
                     const float* be, float* out, int C) {
  extern __shared__ float sh[];
  long long row = blockIdx.x; int c = threadIdx.x;
  float v = A[row*C + c] + (Badd ? Badd[row*C + c] : 0.f);
  sh[c] = v; __syncthreads();
  for (int o = C >> 1; o > 0; o >>= 1) { if (c < o) sh[c] += sh[c+o]; __syncthreads(); }
  float mu = sh[0] / C; __syncthreads();
  float d = v - mu;
  sh[c] = d * d; __syncthreads();
  for (int o = C >> 1; o > 0; o >>= 1) { if (c < o) sh[c] += sh[c+o]; __syncthreads(); }
  float var = sh[0] / C; __syncthreads();
  out[row*C + c] = d * rsqrtf(var + 1e-5f) * g[c] + be[c];
}

__global__ void k_vel(const float* msg, const float* posm, const int* nm, float* vel) {
  int bi = blockIdx.x; int j = threadIdx.x;  // 1024 blocks, 128 threads
  int b = bi >> 7;
  __shared__ float sh[128];
  float mg = msg[(long long)bi*128 + j];
  int q = b*128 + j;
  float comp[3];
  comp[0] = mg * (posm[bi*3]   - posm[q*3]);
  comp[1] = mg * (posm[bi*3+1] - posm[q*3+1]);
  comp[2] = mg * (posm[bi*3+2] - posm[q*3+2]);
  float mi_ = nm[bi] ? 1.f : 0.f;
  for (int k = 0; k < 3; k++) {
    sh[j] = comp[k]; __syncthreads();
    for (int o = 64; o > 0; o >>= 1) { if (j < o) sh[j] += sh[j+o]; __syncthreads(); }
    if (j == 0) vel[bi*3 + k] = sh[0] * mi_;
    __syncthreads();
  }
}

__global__ void k_pos_finish(const float* vel, const float* posm, const int* nm,
                             const float* se3w, float* out) {
  int b = blockIdx.x; int i = threadIdx.x;  // 8 blocks, 128 threads
  __shared__ float sh[128];
  int bi = b*128 + i;
  float m = nm[bi] ? 1.f : 0.f;
  float vv[3] = { vel[bi*3], vel[bi*3+1], vel[bi*3+2] };
  sh[i] = m; __syncthreads();
  for (int o = 64; o > 0; o >>= 1) { if (i < o) sh[i] += sh[i+o]; __syncthreads(); }
  float cnt = sh[0]; __syncthreads();
  float mean[3];
  for (int k = 0; k < 3; k++) {
    sh[i] = vv[k]; __syncthreads();
    for (int o = 64; o > 0; o >>= 1) { if (i < o) sh[i] += sh[i+o]; __syncthreads(); }
    mean[k] = sh[0] / cnt; __syncthreads();
  }
  float px = vv[0] - mean[0]*m, py = vv[1] - mean[1]*m, pz = vv[2] - mean[2]*m;
  float nr = sqrtf(px*px + py*py + pz*pz);
  sh[i] = nr; __syncthreads();
  for (int o = 64; o > 0; o >>= 1) { if (i < o) sh[i] += sh[i+o]; __syncthreads(); }
  float mean_norm = sh[0] / cnt;
  float sc = se3w[0] / (mean_norm + 1e-5f);
  out[bi*3]   = posm[bi*3]   + px * sc;
  out[bi*3+1] = posm[bi*3+1] + py * sc;
  out[bi*3+2] = posm[bi*3+2] + pz * sc;
}

// ---------------- host side ----------------
static inline void gemm(hipStream_t st, const float* A, long long sA, int lda,
                        const float* B, long long sB, int ldb, int tb,
                        const float* bias, float* C, long long sC, long long sC2,
                        int bdiv, int ldc, int M, int Nn, int K, int relu, int nb) {
  dim3 g((Nn + GBN - 1) / GBN, (M + GBM - 1) / GBM, nb);
  k_gemm<<<g, dim3(256), 0, st>>>(A, sA, lda, B, sB, ldb, tb, bias,
                                  C, sC, sC2, bdiv, ldc, M, Nn, K, relu);
}
#define EWG(total) dim3(((total) + 255) / 256)

extern "C" void kernel_launch(void* const* d_in, const int* in_sizes, int n_in,
                              void* d_out, int out_size, void* d_ws, size_t ws_size,
                              hipStream_t stream) {
  (void)in_sizes; (void)n_in; (void)out_size; (void)ws_size;
  int ix = 0;
  auto F = [&]() -> const float* { return (const float*)d_in[ix++]; };
  const float* X    = F();
  const float* E    = F();
  const float* yv   = F();
  const float* pos  = F();
  const int*   nm   = (const int*)d_in[ix++];
  const int*   pmm  = (const int*)d_in[ix++];
  const float* ppos = F(); (void)ppos;
  const float* patom = F();
  const float *inE_w=F(),*inE_b=F();
  const float *xem1_w=F(),*xem1_b=F();
  const float *xem2_w=F(),*xem2_b=F();
  const float *ld1_w=F(),*ld1_b=F();
  const float *lnp1_w=F(),*lnp1_b=F();
  const float *lnp2_w=F(),*lnp2_b=F();
  const float *dae_w=F(),*dae_b=F();
  const float *dme_w=F(),*dme_b=F();
  const float *q_w=F(),*q_b=F();
  const float *k_w=F(),*k_b=F();
  const float *v_w=F(),*v_b=F();
  const float *a_w=F();
  const float *out_w=F(),*out_b=F();
  const float *qc_w=F(),*qc_b=F();
  const float *kc_w=F(),*kc_b=F();
  const float *vc_w=F(),*vc_b=F();
  const float *ac_w=F();
  const float *outc_w=F(),*outc_b=F();
  const float *eatt_w=F(),*eatt_b=F();
  const float *patt_w=F(),*patt_b=F();
  const float *exm_w=F(),*exm_b=F();
  const float *pxm_w=F(),*pxm_b=F();
  const float *yem_w=F(),*yem_b=F();
  const float *yea_w=F(),*yea_b=F();
  const float *yxm_w=F(),*yxm_b=F();
  const float *yxa_w=F(),*yxa_b=F();
  const float *yy_w=F(),*yy_b=F();
  const float *xy_w=F(),*xy_b=F();
  const float *ey_w=F(),*ey_b=F();
  const float *dy_w=F(),*dy_b=F();
  const float *ep1_w=F();
  const float *ep2_w=F();
  const float *xo_w=F(),*xo_b=F();
  const float *eo_w=F(),*eo_b=F();
  const float *yo1_w=F(),*yo1_b=F();
  const float *yo2_w=F(),*yo2_b=F();
  const float *lx1_w=F(),*lx1_b=F();
  const float *lx2_w=F(),*lx2_b=F();
  const float *le1_w=F(),*le1_b=F();
  const float *le2_w=F(),*le2_b=F();
  const float *ly1_w=F(),*ly1_b=F();
  const float *ly2_w=F(),*ly2_b=F();
  const float *nx1_g=F(),*nx1_b=F();
  const float *nx2_g=F(),*nx2_b=F();
  const float *ne1_g=F(),*ne1_b=F();
  const float *ne2_g=F(),*ne2_b=F();
  const float *ny1_g=F(),*ny1_b=F();
  const float *ny2_g=F(),*ny2_b=F();
  const float *se3w=F();

  // output regions (X_out, E_out, y_out, pos_out)
  float* out_X = (float*)d_out;
  float* out_E = out_X + NN_ * C_DX;
  float* out_y = out_E + (long long)NE * C_DE;
  float* out_p = out_y + C_BS * 64;

  // workspace bump allocator
  char* wp = (char*)d_ws;
  auto alloc = [&](size_t nfl) -> float* {
    float* p = (float*)wp;
    wp += ((nfl * 4 + 255) / 256) * 256;
    return p;
  };
  float* posm  = alloc(NN_*3);
  float* npos  = alloc(NN_*3);
  float* normp = alloc(NN_);
  float* pinfo = alloc((size_t)NE*2);
  float* eA  = alloc((size_t)NE*64);      // dist1
  float* eB  = alloc((size_t)NE*64);      // Y
  float* eC  = alloc((size_t)NE*64);      // E2
  float* eD2 = alloc((size_t)NE*64);      // Eout_raw -> ffnE2
  float* eE  = alloc((size_t)NE*64);      // Er
  float* eBig = alloc((size_t)NE*128);    // dA|dM -> messages -> ffnE1
  float* dA = eBig; float* dM = eBig + (size_t)NE*64;
  float* msg  = alloc(NE);
  float* xm1 = alloc(NN_*64); float* xm2 = alloc(NN_*64);
  float* yadd = alloc(512); float* ymul = alloc(512);
  float* Qb = alloc(NN_*256); float* Kb = alloc(NN_*256); float* Vb = alloc(NN_*256);
  float* Qcb = alloc(NN_*256); float* Kcb = alloc(NN_*256); float* Vcb = alloc(NN_*256);
  float* Qh  = alloc((size_t)C_BS*C_NH*C_N*C_DX);
  float* Qch = alloc((size_t)C_BS*C_NH*C_N*C_DX);
  float* eam = alloc((size_t)NE*8); float* pam = alloc((size_t)NE*8);
  float* sbase = alloc((size_t)64*16384); float* sm  = alloc((size_t)64*16384);
  float* scb   = alloc((size_t)64*16384); float* smc = alloc((size_t)64*16384);
  float* maxv = alloc(NE);
  float* wsum = alloc(8);
  float* wVcat = alloc((size_t)NN_*2048); float* wVccat = alloc((size_t)NN_*2048);
  float* wV = alloc(NN_*256); float* wVc = alloc(NN_*256);
  float* Rex = alloc(NN_*256); float* Rpx = alloc(NN_*256);
  float* bexm = alloc(NN_*256); float* bpxm = alloc(NN_*256);
  float* byxm = alloc(2048); float* byxa = alloc(2048);
  float* newX = alloc(NN_*256);
  float* XoutR = alloc(NN_*256); float* Xr = alloc(NN_*256);
  float* fX1 = alloc((size_t)NN_*2048); float* fX2 = alloc(NN_*256);
  float* RyY = alloc(2048); float* RyD = alloc(2048); float* Rx = alloc(8192);
  float* newy = alloc(512); float* t1 = alloc(512); float* t2 = alloc(512);
  float* yr = alloc(512); float* fy1 = alloc(16384); float* fy2 = alloc(512);
  float* vel = alloc(NN_*3);

  const float inv_sqrt_de = 0.125f;  // 1/sqrt(DE=64)

  // ---- geometry + dist1 ----
  k_prep_pos<<<EWG(NN_), 256, 0, stream>>>(pos, nm, posm, npos, normp);
  k_pinfo<<<EWG(NE), 256, 0, stream>>>(posm, npos, pinfo);
  k_dist1<<<EWG(NE*64), 256, 0, stream>>>(pinfo, normp, nm, ld1_w, ld1_b,
                                          lnp1_w, lnp1_b, lnp2_w, lnp2_b, eA);
  // ---- edge features Y / E2 ----
  gemm(stream, X,0,256, xem1_w,0,64,0, xem1_b, xm1,0,0,1,64, NN_,64,256,0,1);
  k_maskx<<<EWG(NN_*64), 256, 0, stream>>>(xm1, nm, 64, NN_*64);
  gemm(stream, X,0,256, xem2_w,0,64,0, xem2_b, xm2,0,0,1,64, NN_,64,256,0,1);
  k_maskx<<<EWG(NN_*64), 256, 0, stream>>>(xm2, nm, 64, NN_*64);
  gemm(stream, E,0,64, inE_w,0,64,0, inE_b, eB,0,0,1,64, NE,64,64,0,1);
  gemm(stream, eA,0,64, dae_w,0,64,0, dae_b, dA,0,0,1,64, NE,64,64,0,1);
  gemm(stream, eA,0,64, dme_w,0,64,0, dme_b, dM,0,0,1,64, NE,64,64,0,1);
  k_small_lin<<<EWG(512), 256, 0, stream>>>(yv,64, yea_w, yea_b, yadd,64, 8,64,64,0,0);
  k_small_lin<<<EWG(512), 256, 0, stream>>>(yv,64, yem_w, yem_b, ymul,64, 8,64,64,0,0);
  k_combineE<<<EWG(NE*64), 256, 0, stream>>>(eB, xm1, xm2, dA, dM, yadd, ymul, nm, eB, eC);

  // ---- position messages / velocity (frees eBig afterwards) ----
  gemm(stream, eB,0,64, ep1_w,0,64,0, nullptr, eBig,0,0,1,64, NE,64,64,1,1);
  gemm(stream, eBig,0,64, ep2_w,0,1,0, nullptr, msg,0,0,1,1, NE,1,64,0,1);
  k_vel<<<dim3(NN_), dim3(128), 0, stream>>>(msg, posm, nm, vel);
  k_pos_finish<<<dim3(C_BS), dim3(128), 0, stream>>>(vel, posm, nm, se3w, out_p);

  // ---- E output branch ----
  gemm(stream, eC,0,64, eo_w,0,64,0, eo_b, eD2,0,0,1,64, NE,64,64,0,1);
  k_maske<<<EWG(NE*64), 256, 0, stream>>>(eD2, nm, 64);
  k_ln<<<dim3(NE), dim3(64), 64*4, stream>>>(E, eD2, ne1_g, ne1_b, eE, 64);

  // ---- self attention ----
  gemm(stream, eC,0,64, eatt_w,0,8,0, eatt_b, eam,0,0,1,8, NE,8,64,0,1);
  gemm(stream, eA,0,64, patt_w,0,8,0, patt_b, pam,0,0,1,8, NE,8,64,0,1);
  gemm(stream, X,0,256, q_w,0,256,0, q_b, Qb,0,0,1,256, NN_,256,256,0,1);
  k_maskx<<<EWG(NN_*256), 256, 0, stream>>>(Qb, nm, 256, NN_*256);
  gemm(stream, X,0,256, k_w,0,256,0, k_b, Kb,0,0,1,256, NN_,256,256,0,1);
  k_maskx<<<EWG(NN_*256), 256, 0, stream>>>(Kb, nm, 256, NN_*256);
  gemm(stream, X,0,256, v_w,0,256,0, v_b, Vb,0,0,1,256, NN_,256,256,0,1);
  k_maskx<<<EWG(NN_*256), 256, 0, stream>>>(Vb, nm, 256, NN_*256);
  k_make_qh<<<EWG(C_BS*C_NH*C_N*C_DX), 256, 0, stream>>>(Qb, a_w, Qh, inv_sqrt_de);
  gemm(stream, Qh,32768,256, Kb,32768,256,1, nullptr,
       sbase,131072,16384,8, 128, 128,128,256,0, 64);
  k_score_self<<<EWG(64*16384), 256, 0, stream>>>(sbase, eam, pam, nm, sm);
  k_softmax<<<dim3(64*128), dim3(128), 0, stream>>>(sm, nm);
  gemm(stream, sm,16384,128, Vb,32768,256,0, nullptr,
       wVcat,262144,256,8, 2048, 128,256,128,0, 64);
  gemm(stream, wVcat,0,2048, out_w,0,256,0, out_b, wV,0,0,1,256, NN_,256,2048,0,1);
  k_maskx<<<EWG(NN_*256), 256, 0, stream>>>(wV, nm, 256, NN_*256);

  // ---- cross attention ----
  gemm(stream, wV,0,256, qc_w,0,256,0, qc_b, Qcb,0,0,1,256, NN_,256,256,0,1);
  k_maskx<<<EWG(NN_*256), 256, 0, stream>>>(Qcb, pmm, 256, NN_*256);
  gemm(stream, patom,0,256, kc_w,0,256,0, kc_b, Kcb,0,0,1,256, NN_,256,256,0,1);
  k_maskx<<<EWG(NN_*256), 256, 0, stream>>>(Kcb, pmm, 256, NN_*256);
  gemm(stream, patom,0,256, vc_w,0,256,0, vc_b, Vcb,0,0,1,256, NN_,256,256,0,1);
  k_maskx<<<EWG(NN_*256), 256, 0, stream>>>(Vcb, pmm, 256, NN_*256);
  k_make_qh<<<EWG(C_BS*C_NH*C_N*C_DX), 256, 0, stream>>>(Qcb, ac_w, Qch, inv_sqrt_de);
  gemm(stream, Qch,32768,256, Kcb,32768,256,1, nullptr,
       scb,131072,16384,8, 128, 128,128,256,0, 64);
  k_cross_max<<<dim3(NN_), dim3(128), 0, stream>>>(Qcb, Kcb, maxv, inv_sqrt_de);
  k_wsum<<<dim3(1), dim3(8), 0, stream>>>(ac_w, wsum);
  k_score_cross<<<EWG(64*16384), 256, 0, stream>>>(scb, maxv, wsum, pmm, smc);
  k_softmax<<<dim3(64*128), dim3(128), 0, stream>>>(smc, pmm);
  gemm(stream, smc,16384,128, Vcb,32768,256,0, nullptr,
       wVccat,262144,256,8, 2048, 128,256,128,0, 64);
  gemm(stream, wVccat,0,2048, outc_w,0,256,0, outc_b, wVc,0,0,1,256, NN_,256,2048,0,1);
  k_maskx<<<EWG(NN_*256), 256, 0, stream>>>(wVc, pmm, 256, NN_*256);

  // ---- edge->node modulations, newX ----
  k_reduce_x<<<dim3(NN_), dim3(64), 0, stream>>>(eC, nm, Rex);
  gemm(stream, Rex,0,256, exm_w,0,256,0, exm_b, bexm,0,0,1,256, NN_,256,256,0,1);
  k_reduce_x<<<dim3(NN_), dim3(64), 0, stream>>>(eA, nm, Rpx);
  gemm(stream, Rpx,0,256, pxm_w,0,256,0, pxm_b, bpxm,0,0,1,256, NN_,256,256,0,1);
  k_small_lin<<<EWG(8*256), 256, 0, stream>>>(yv,64, yxm_w, yxm_b, byxm,256, 8,256,64,0,0);
  k_small_lin<<<EWG(8*256), 256, 0, stream>>>(yv,64, yxa_w, yxa_b, byxa,256, 8,256,64,0,0);
  k_newx<<<EWG(NN_*256), 256, 0, stream>>>(wV, wVc, pmm, bexm, bpxm, byxm, byxa, newX);

  // ---- X output branch ----
  gemm(stream, newX,0,256, xo_w,0,256,0, xo_b, XoutR,0,0,1,256, NN_,256,256,0,1);
  k_maskx<<<EWG(NN_*256), 256, 0, stream>>>(XoutR, nm, 256, NN_*256);
  k_ln<<<dim3(NN_), dim3(256), 256*4, stream>>>(X, XoutR, nx1_g, nx1_b, Xr, 256);
  gemm(stream, Xr,0,256, lx1_w,0,2048,0, lx1_b, fX1,0,0,1,2048, NN_,2048,256,1,1);
  gemm(stream, fX1,0,2048, lx2_w,0,256,0, lx2_b, fX2,0,0,1,256, NN_,256,2048,0,1);
  k_ln<<<dim3(NN_), dim3(256), 256*4, stream>>>(Xr, fX2, nx2_g, nx2_b, out_X, 256);

  // ---- E FFN + final LN (eBig reused for linE1 activations) ----
  gemm(stream, eE,0,64, le1_w,0,128,0, le1_b, eBig,0,0,1,128, NE,128,64,1,1);
  gemm(stream, eBig,0,128, le2_w,0,64,0, le2_b, eD2,0,0,1,64, NE,64,128,0,1);
  k_ln<<<dim3(NE), dim3(64), 64*4, stream>>>(eE, eD2, ne2_g, ne2_b, out_E, 64);

  // ---- y path ----
  k_reduce_y<<<dim3(C_BS), dim3(64), 0, stream>>>(eB, nm, RyY);
  k_reduce_y<<<dim3(C_BS), dim3(64), 0, stream>>>(eA, nm, RyD);
  k_reduce_x2y<<<dim3(C_BS), dim3(256), 0, stream>>>(newX, nm, Rx);
  k_small_lin<<<EWG(512), 256, 0, stream>>>(yv,64, yy_w, yy_b, newy,64, 8,64,64,0,0);
  k_small_lin<<<EWG(512), 256, 0, stream>>>(Rx,1024, xy_w, xy_b, newy,64, 8,64,1024,0,1);
  k_small_lin<<<EWG(512), 256, 0, stream>>>(RyY,256, ey_w, ey_b, newy,64, 8,64,256,0,1);
  k_small_lin<<<EWG(512), 256, 0, stream>>>(RyD,256, dy_w, dy_b, newy,64, 8,64,256,0,1);
  k_small_lin<<<EWG(512), 256, 0, stream>>>(newy,64, yo1_w, yo1_b, t1,64, 8,64,64,1,0);
  k_small_lin<<<EWG(512), 256, 0, stream>>>(t1,64, yo2_w, yo2_b, t2,64, 8,64,64,0,0);
  k_ln<<<dim3(C_BS), dim3(64), 64*4, stream>>>(yv, t2, ny1_g, ny1_b, yr, 64);
  k_small_lin<<<EWG(8*2048), 256, 0, stream>>>(yr,64, ly1_w, ly1_b, fy1,2048, 8,2048,64,1,0);
  k_small_lin<<<EWG(512), 256, 0, stream>>>(fy1,2048, ly2_w, ly2_b, fy2,64, 8,64,2048,0,0);
  k_ln<<<dim3(C_BS), dim3(64), 64*4, stream>>>(yr, fy2, ny2_g, ny2_b, out_y, 64);
}
